// CollabFFLayer_33732673143833
// MI455X (gfx1250) — matrix operand order repaired and checked by reading, not compiled, not run on hardware
//
#include <hip/hip_runtime.h>

// Problem constants (from reference): x[8192,4096] fp32, W[4096,4096] fp32, b[4096] fp32
#define B_ROWS 8192
#define K_DIM  4096
#define N_COLS 4096

// GEMM staging geometry
#define KB         64                    // K elements staged per block (128 B per row)
#define NKB        (K_DIM / KB)          // 64 K-blocks
#define TILE_BYTES (128 * 128)           // 128 rows x 128 B = 16 KB per tile
#define BUF_BYTES  (2 * TILE_BYTES)      // A tile + B tile = 32 KB per buffer

typedef __attribute__((ext_vector_type(16))) __bf16 v16bf;
typedef __attribute__((ext_vector_type(8)))  float  v8f;

__device__ __forceinline__ unsigned short f32_to_bf16(float f) {
  unsigned int u = __float_as_uint(f);
  u += 0x7FFFu + ((u >> 16) & 1u);           // round-to-nearest-even
  return (unsigned short)(u >> 16);
}

// ---------------------------------------------------------------------------
// Kernel 1: per-row L2 normalize x (fp32) and emit bf16 rows into workspace.
// ---------------------------------------------------------------------------
__global__ void __launch_bounds__(256) rownorm_to_bf16(const float* __restrict__ x,
                                                       unsigned short* __restrict__ xn) {
  const int row = blockIdx.x;
  const float4* xr = (const float4*)(x + (size_t)row * K_DIM);

  float s = 0.f;
#pragma unroll
  for (int it = 0; it < (K_DIM / 4) / 256; ++it) {
    float4 v = xr[it * 256 + threadIdx.x];
    s += v.x * v.x + v.y * v.y + v.z * v.z + v.w * v.w;
  }
#pragma unroll
  for (int off = 16; off > 0; off >>= 1) s += __shfl_xor(s, off, 32);

  __shared__ float sm[8];
  const int wave = threadIdx.x >> 5;
  const int lane = threadIdx.x & 31;
  if (lane == 0) sm[wave] = s;
  __syncthreads();
  const float tot = sm[0] + sm[1] + sm[2] + sm[3] + sm[4] + sm[5] + sm[6] + sm[7];
  const float scale = 1.f / (sqrtf(tot) + 1e-8f);

  unsigned short* xo = xn + (size_t)row * K_DIM;
#pragma unroll
  for (int it = 0; it < (K_DIM / 4) / 256; ++it) {
    const int i = it * 256 + threadIdx.x;
    float4 v = xr[i];
    ushort4 o;
    o.x = f32_to_bf16(v.x * scale);
    o.y = f32_to_bf16(v.y * scale);
    o.z = f32_to_bf16(v.z * scale);
    o.w = f32_to_bf16(v.w * scale);
    *(ushort4*)(xo + 4 * (size_t)i) = o;
  }
}

// ---------------------------------------------------------------------------
// Kernel 2: W fp32 -> bf16 (grid-stride, float4 loads / ushort4 stores)
// ---------------------------------------------------------------------------
__global__ void __launch_bounds__(256) w_to_bf16(const float* __restrict__ w,
                                                 unsigned short* __restrict__ wb) {
  const size_t n4 = (size_t)N_COLS * K_DIM / 4;
  const size_t stride = (size_t)gridDim.x * blockDim.x;
  for (size_t i = (size_t)blockIdx.x * blockDim.x + threadIdx.x; i < n4; i += stride) {
    float4 v = ((const float4*)w)[i];
    ushort4 o;
    o.x = f32_to_bf16(v.x);
    o.y = f32_to_bf16(v.y);
    o.z = f32_to_bf16(v.z);
    o.w = f32_to_bf16(v.w);
    ((ushort4*)wb)[i] = o;
  }
}

// ---------------------------------------------------------------------------
// Kernel 3: bf16 WMMA GEMM with CDNA5 async global->LDS staging.
//   out = relu(xn @ W^T + b), fp32 accumulate.
// Block = 8 waves -> 128(M) x 128(N); wave = 32(M) x 64(N) = 2x4 WMMA tiles.
// Tiles staged 64-K at a time into LDS (double buffered, 64 KB total) via
// global_load_async_to_lds_b128 (ASYNCcnt), fragments read via ds_load_b128.
// LDS rows are 128 B; 16 B chunk c of row r stored at chunk (c ^ ((r>>1)&7))
// -> every 32-lane ds_load_b128 fragment read covers all 64 banks twice
//    (2 phases, conflict-free).
// ---------------------------------------------------------------------------
__global__ void __launch_bounds__(256) gemm_bf16_wmma(const unsigned short* __restrict__ xn,
                                                      const unsigned short* __restrict__ wb,
                                                      const float* __restrict__ bias,
                                                      float* __restrict__ out) {
  extern __shared__ unsigned char smem[];   // 2 * 32 KB, passed at launch

  const int tid   = threadIdx.x;
  const int lane  = tid & 31;
  const int wave  = tid >> 5;
  const int lhalf = lane >> 4;      // 0 or 1
  const int lmod  = lane & 15;
  const int waveM = wave >> 1;      // 0..3
  const int waveN = wave & 1;       // 0..1

  const int rowBlk = blockIdx.y * 128;               // block M base
  const int colBlk = blockIdx.x * 128;               // block N base
  const int rowBase = rowBlk + waveM * 32;           // wave M base
  const int colBase = colBlk + waveN * 64;           // wave N base

  // LDS byte offset of dynamic shared (generic VA low 32 bits == LDS offset)
  const unsigned ldsBase = (unsigned)(size_t)(&smem[0]);

  // ---- staging map: thread handles flat chunks f = tid + 256*j (j=0..3) of
  // each 1024-chunk tile; row = f>>3, chunk = f&7. One wave instruction then
  // covers 4 complete, contiguous 128 B rows (fully coalesced).
  unsigned aLdsOff[4], bLdsOff[4];
  const unsigned short* aGlob[4];
  const unsigned short* bGlob[4];
#pragma unroll
  for (int j = 0; j < 4; ++j) {
    const int f = tid + 256 * j;
    const int r = f >> 3;
    const int c = f & 7;
    const unsigned sw = (unsigned)((c ^ ((r >> 1) & 7)) * 16);
    aLdsOff[j] = (unsigned)(r * 128) + sw;
    bLdsOff[j] = (unsigned)(TILE_BYTES + r * 128) + sw;
    aGlob[j] = xn + (size_t)(rowBlk + r) * K_DIM + c * 8;
    bGlob[j] = wb + (size_t)(colBlk + r) * K_DIM + c * 8;
  }

  // issue this thread's 8 async 16 B copies for K-block kb into buffer buf
  auto stage = [&](int buf, int kb) {
    const unsigned bofs = ldsBase + (unsigned)(buf * BUF_BYTES);
    const size_t kofs = (size_t)kb * KB;     // 64 bf16 = 128 B per row
#pragma unroll
    for (int j = 0; j < 4; ++j) {
      unsigned long long ga = (unsigned long long)(size_t)(aGlob[j] + kofs);
      asm volatile("global_load_async_to_lds_b128 %0, %1, off"
                   :: "v"(bofs + aLdsOff[j]), "v"(ga) : "memory");
    }
#pragma unroll
    for (int j = 0; j < 4; ++j) {
      unsigned long long gb = (unsigned long long)(size_t)(bGlob[j] + kofs);
      asm volatile("global_load_async_to_lds_b128 %0, %1, off"
                   :: "v"(bofs + bLdsOff[j]), "v"(gb) : "memory");
    }
  };

  // ---- per-lane LDS fragment read bases (offsets relative to smem[])
  unsigned aRowL[2]; int aSw[2];
#pragma unroll
  for (int mt = 0; mt < 2; ++mt) {
    const int rT = waveM * 32 + mt * 16 + lmod;      // row within A tile
    aRowL[mt] = (unsigned)(rT * 128);
    aSw[mt] = (rT >> 1) & 7;
  }
  unsigned bRowL[4]; int bSw[4];
#pragma unroll
  for (int nt = 0; nt < 4; ++nt) {
    const int rT = waveN * 64 + nt * 16 + lmod;      // row within B tile
    bRowL[nt] = (unsigned)(TILE_BYTES + rT * 128);
    bSw[nt] = (rT >> 1) & 7;
  }

  union Frag { uint4 u[2]; v16bf v; };

  v8f zero = {};
  v8f acc[2][4];
#pragma unroll
  for (int mt = 0; mt < 2; ++mt)
#pragma unroll
    for (int nt = 0; nt < 4; ++nt) acc[mt][nt] = zero;

  // ---- prologue: fill buffer 0
  stage(0, 0);
  asm volatile("s_wait_asynccnt 0" ::: "memory");
  __syncthreads();

  for (int kb = 0; kb < NKB; ++kb) {
    const int cur = kb & 1;
    if (kb + 1 < NKB) stage(cur ^ 1, kb + 1);        // overlap copy with compute

    const unsigned bofs = (unsigned)(cur * BUF_BYTES);
#pragma unroll
    for (int kk = 0; kk < 2; ++kk) {                 // two 32-K WMMA steps
      Frag A[2], Bf[4];
#pragma unroll
      for (int mt = 0; mt < 2; ++mt) {
        const int c0 = kk * 4 + lhalf;               // K: +h*8..+7
        const int c1 = c0 + 2;                       // K: +16+h*8..+7
        A[mt].u[0] = *(const uint4*)(smem + bofs + aRowL[mt] + ((c0 ^ aSw[mt]) * 16));
        A[mt].u[1] = *(const uint4*)(smem + bofs + aRowL[mt] + ((c1 ^ aSw[mt]) * 16));
      }
#pragma unroll
      for (int nt = 0; nt < 4; ++nt) {
        const int c0 = kk * 4 + lhalf;               // K: +h*8..+7
        const int c1 = c0 + 2;                       // K: +16+h*8..+7
        Bf[nt].u[0] = *(const uint4*)(smem + bofs + bRowL[nt] + ((c0 ^ bSw[nt]) * 16));
        Bf[nt].u[1] = *(const uint4*)(smem + bofs + bRowL[nt] + ((c1 ^ bSw[nt]) * 16));
      }
#pragma unroll
      for (int mt = 0; mt < 2; ++mt)
#pragma unroll
        for (int nt = 0; nt < 4; ++nt)
          acc[mt][nt] = __builtin_amdgcn_wmma_f32_16x16x32_bf16(
              false, A[mt].v, false, Bf[nt].v, (short)0, acc[mt][nt], false, false);
    }

    // this wave's async copies done; barrier makes all waves' copies visible
    // and protects the buffer we are about to overwrite next iteration
    asm volatile("s_wait_asynccnt 0" ::: "memory");
    __syncthreads();
  }

  // ---- epilogue: C/D layout -> vgpr i: lanes 0-15 M=i, lanes 16-31 M=i+8; N = lane&15
  const int mHalf = lhalf * 8;
#pragma unroll
  for (int nt = 0; nt < 4; ++nt) {
    const int col = colBase + nt * 16 + lmod;
    const float bv = bias[col];
#pragma unroll
    for (int mt = 0; mt < 2; ++mt) {
      const int rowT = rowBase + mt * 16 + mHalf;
#pragma unroll
      for (int i = 0; i < 8; ++i) {
        float v = acc[mt][nt][i] + bv;
        v = v > 0.f ? v : 0.f;
        out[(size_t)(rowT + i) * N_COLS + col] = v;
      }
    }
  }
}

// ---------------------------------------------------------------------------
extern "C" void kernel_launch(void* const* d_in, const int* in_sizes, int n_in,
                              void* d_out, int out_size, void* d_ws, size_t ws_size,
                              hipStream_t stream) {
  const float* x = (const float*)d_in[0];
  const float* W = (const float*)d_in[1];
  const float* b = (const float*)d_in[2];
  float* out = (float*)d_out;

  // workspace layout: [xn bf16: 64 MiB][W bf16: 32 MiB]
  unsigned short* xn  = (unsigned short*)d_ws;
  unsigned short* wbf = xn + (size_t)B_ROWS * K_DIM;

  rownorm_to_bf16<<<B_ROWS, 256, 0, stream>>>(x, xn);
  w_to_bf16<<<4096, 256, 0, stream>>>(W, wbf);

  dim3 grid(N_COLS / 128, B_ROWS / 128);  // (32, 64)
  gemm_bf16_wmma<<<grid, 256, 2 * BUF_BYTES, stream>>>(xn, wbf, b, out);
}
